// LatentCorrelationLayer_25744033972833
// MI455X (gfx1250) — compile-verified, matrix operand order checked by reading, not run on hardware
//
#include <hip/hip_runtime.h>
#include <hip/hip_bf16.h>

typedef __attribute__((ext_vector_type(16))) _Float16 v16h;
typedef __attribute__((ext_vector_type(8)))  _Float16 v8h;
typedef __attribute__((ext_vector_type(8)))  float    v8f;

constexpr int  B_ = 32, T_ = 2048, N_ = 512, H_ = 1024, O_ = 512;
constexpr long long M_ = (long long)B_ * T_;          // 65536 rows of [M, *]
constexpr float EPS_ = 1e-8f;
constexpr float LN_EPS_ = 1e-5f;

// ---------------- workspace arena (bytes) ----------------
constexpr size_t XLN_OFF  = 0;                                   // f32 [M,N]   128MB (reused as HH f16 [M,H])
constexpr size_t XH_OFF   = XLN_OFF  + (size_t)M_ * N_ * 4;      // f16 [M,N]    64MB (reused as OUTH f16)
constexpr size_t XNT_OFF  = XH_OFF   + (size_t)M_ * N_ * 2;      // f16 [B,N,T]  64MB (reused as TX1H f16)
constexpr size_t HPRE_OFF = XNT_OFF  + (size_t)B_ * N_ * T_ * 2; // f32 [M,H]   256MB
constexpr size_t WIH_OFF  = HPRE_OFF + (size_t)M_ * H_ * 4;      // f16 [H,N]
constexpr size_t WOH_OFF  = WIH_OFF  + (size_t)H_ * N_ * 2;      // f16 [O,H]
constexpr size_t LHT_OFF  = WOH_OFF  + (size_t)O_ * H_ * 2;      // f16 [N,N] (transposed L)
constexpr size_t CORR_OFF = LHT_OFF  + (size_t)N_ * N_ * 2;      // f32 [N,N]
constexpr size_t A0_OFF   = CORR_OFF + (size_t)N_ * N_ * 4;      // f32 [N,N]
constexpr size_t MEAN_OFF = A0_OFF   + (size_t)N_ * N_ * 4;      // f32 [B,N]
constexpr size_t INVN_OFF = MEAN_OFF + (size_t)B_ * N_ * 4;      // f32 [B,N]
constexpr size_t DIS_OFF  = INVN_OFF + (size_t)B_ * N_ * 4;      // f32 [N]

// ---------------- GEMM tiling ----------------
// Block tile 128x128, 8 waves (4 along M, 2 along N), each wave 32x64 = 2x4 WMMA tiles.
// LDS double-buffered with a manually unrolled ping-pong (steps must be even).
constexpr int BM = 128, BN = 128, BK = 32;
constexpr int LDK_S = BK + 8;            // 40 halfs = 80B rows (16B aligned)

// ---------------- helpers ----------------
__device__ __forceinline__ float block_reduce_sum256(float v, float* sd) {
  int tid = threadIdx.x;
  sd[tid] = v; __syncthreads();
  #pragma unroll
  for (int s = 128; s > 0; s >>= 1) {
    if (tid < s) sd[tid] += sd[tid + s];
    __syncthreads();
  }
  float r = sd[0]; __syncthreads();
  return r;
}

__device__ __forceinline__ v8f wmma_f16(v16h a, v16h b, v8f c) {
  // D = A(16x32 f16) * B(32x16 f16) + C(16x16 f32)
  return __builtin_amdgcn_wmma_f32_16x16x32_f16(false, a, false, b, (short)0, c, false, false);
}

// K-contiguous fragment: lane&15 selects row (A: M-row, B: N-column);
// lane>>4 selects K halves {0..7,16..23} vs {8..15,24..31}.
// Each half is a contiguous, 16B-aligned 8xf16 run -> 2x ds_load_b128.
__device__ __forceinline__ v16h frag16(const _Float16 (*S)[LDK_S], int base, int lane) {
  int idx = base + (lane & 15);
  int hi  = (lane >> 4) & 1;
  v8h lo = *(const v8h*)&S[idx][8 * hi];
  v8h hv = *(const v8h*)&S[idx][16 + 8 * hi];
  v16h r;
  #pragma unroll
  for (int i = 0; i < 8; ++i) { r[i] = lo[i]; r[8 + i] = hv[i]; }
  return r;
}

__device__ __forceinline__ float clipf(float v, float lo, float hi) {
  return fminf(fmaxf(v, lo), hi);
}

// stage one K-slice (this thread's 4x 16B chunks) into one LDS buffer
__device__ __forceinline__ void stage_tile(_Float16 (*Asb)[LDK_S], _Float16 (*Bsb)[LDK_S],
                                           int r0, int r1, int c0,
                                           const _Float16* a0, const _Float16* a1,
                                           const _Float16* b0, const _Float16* b1) {
  *(v8h*)&Asb[r0][c0] = *(const v8h*)a0;
  *(v8h*)&Asb[r1][c0] = *(const v8h*)a1;
  *(v8h*)&Bsb[r0][c0] = *(const v8h*)b0;
  *(v8h*)&Bsb[r1][c0] = *(const v8h*)b1;
}

// 8 WMMAs on one LDS buffer (wave tile 32x64)
__device__ __forceinline__ void mma_step(const _Float16 (*Asb)[LDK_S],
                                         const _Float16 (*Bsb)[LDK_S],
                                         int wm, int wn, int lane, v8f acc[2][4]) {
  v16h fb0 = frag16(Bsb, wn * 64,      lane);
  v16h fb1 = frag16(Bsb, wn * 64 + 16, lane);
  v16h fb2 = frag16(Bsb, wn * 64 + 32, lane);
  v16h fb3 = frag16(Bsb, wn * 64 + 48, lane);
  {
    v16h fa0 = frag16(Asb, wm * 32, lane);
    acc[0][0] = wmma_f16(fa0, fb0, acc[0][0]);
    acc[0][1] = wmma_f16(fa0, fb1, acc[0][1]);
    acc[0][2] = wmma_f16(fa0, fb2, acc[0][2]);
    acc[0][3] = wmma_f16(fa0, fb3, acc[0][3]);
  }
  {
    v16h fa1 = frag16(Asb, wm * 32 + 16, lane);
    acc[1][0] = wmma_f16(fa1, fb0, acc[1][0]);
    acc[1][1] = wmma_f16(fa1, fb1, acc[1][1]);
    acc[1][2] = wmma_f16(fa1, fb2, acc[1][2]);
    acc[1][3] = wmma_f16(fa1, fb3, acc[1][3]);
  }
}

// Shared GEMM core: C += A[M,K] * Bt[N,K]^T over nbatch batches.
// Per-thread persistent source pointers (stride BK per slice); LDS is
// double-buffered with a manually unrolled ping-pong so `buf` is a
// compile-time constant (keeps accumulators in fixed register homes).
// Requires nbatch*(K/BK) even (true for all call sites here).
__device__ __forceinline__ void gemm_core(const _Float16* __restrict__ Ag, long long ldA,
                                          const _Float16* __restrict__ Bg, long long ldB,
                                          int K, int nbatch,
                                          long long bStrideA, long long bStrideB,
                                          int rowBase, int colBase,
                                          _Float16 (*As)[BM][LDK_S],
                                          _Float16 (*Bs)[BN][LDK_S],
                                          v8f acc[2][4]) {
  const int tid  = threadIdx.x;
  const int lane = tid & 31;
  const int wave = tid >> 5;
  const int wm = wave & 3;        // 4 waves along M, 32 rows each
  const int wn = wave >> 2;       // 2 waves along N, 64 cols each

  // this thread's two 16B staging chunks for A and for B
  const int r0 = tid >> 2;                 // rows 0..63
  const int r1 = r0 + 64;                  // rows 64..127
  const int c0 = (tid & 3) << 3;           // half-offset within row
  const _Float16* a0 = Ag + (long long)(rowBase + r0) * ldA + c0;
  const _Float16* a1 = Ag + (long long)(rowBase + r1) * ldA + c0;
  const _Float16* b0 = Bg + (long long)(colBase + r0) * ldB + c0;
  const _Float16* b1 = Bg + (long long)(colBase + r1) * ldB + c0;

  const int kSteps = K / BK;
  const int steps  = nbatch * kSteps;
  const long long advA_b = bStrideA - (long long)(kSteps - 1) * BK;  // batch-boundary advance
  const long long advB_b = bStrideB - (long long)(kSteps - 1) * BK;

  // stage slice 0 into buffer 0, advance to slice 1
  stage_tile(As[0], Bs[0], r0, r1, c0, a0, a1, b0, b1);
  {
    long long aAdv = (1 % kSteps == 0) ? advA_b : (long long)BK;
    long long bAdv = (1 % kSteps == 0) ? advB_b : (long long)BK;
    a0 += aAdv; a1 += aAdv; b0 += bAdv; b1 += bAdv;
  }
  __syncthreads();

  for (int s = 0; s < steps; s += 2) {
    // stage slice s+1 into buffer 1 while computing slice s from buffer 0
    if (s + 1 < steps) {
      stage_tile(As[1], Bs[1], r0, r1, c0, a0, a1, b0, b1);
      __builtin_prefetch(a0 + BK, 0, 1);
      __builtin_prefetch(b0 + BK, 0, 1);
      long long aAdv = ((s + 2) % kSteps == 0) ? advA_b : (long long)BK;
      long long bAdv = ((s + 2) % kSteps == 0) ? advB_b : (long long)BK;
      a0 += aAdv; a1 += aAdv; b0 += bAdv; b1 += bAdv;
    }
    mma_step(As[0], Bs[0], wm, wn, lane, acc);
    __syncthreads();
    // stage slice s+2 into buffer 0 while computing slice s+1 from buffer 1
    if (s + 1 < steps) {
      if (s + 2 < steps) {
        stage_tile(As[0], Bs[0], r0, r1, c0, a0, a1, b0, b1);
        __builtin_prefetch(a0 + BK, 0, 1);
        __builtin_prefetch(b0 + BK, 0, 1);
        long long aAdv = ((s + 3) % kSteps == 0) ? advA_b : (long long)BK;
        long long bAdv = ((s + 3) % kSteps == 0) ? advB_b : (long long)BK;
        a0 += aAdv; a1 += aAdv; b0 += bAdv; b1 += bAdv;
      }
      mma_step(As[1], Bs[1], wm, wn, lane, acc);
      __syncthreads();
    }
  }
}

// ---------------- elementwise / reduction kernels ----------------

__global__ void __launch_bounds__(256) k_cast_f16(const float* __restrict__ s,
                                                  _Float16* __restrict__ d, int n) {
  int i = blockIdx.x * 256 + threadIdx.x;
  if (i < n) d[i] = (_Float16)s[i];
}

// LayerNorm over N=512, one row (b,t) per 256-thread block; writes f32 + f16
__global__ void __launch_bounds__(256) k_ln_x(const float* __restrict__ x,
                                              const float* __restrict__ g,
                                              const float* __restrict__ bta,
                                              float* __restrict__ xln,
                                              _Float16* __restrict__ xh) {
  __shared__ float sd[256];
  long long row = blockIdx.x;
  int c0 = threadIdx.x, c1 = threadIdx.x + 256;
  float v0 = x[row * N_ + c0], v1 = x[row * N_ + c1];
  float mu = block_reduce_sum256(v0 + v1, sd) * (1.0f / N_);
  float d0 = v0 - mu, d1 = v1 - mu;
  float var = block_reduce_sum256(d0 * d0 + d1 * d1, sd) * (1.0f / N_);
  float inv = rsqrtf(var + LN_EPS_);
  float o0 = d0 * inv * g[c0] + bta[c0];
  float o1 = d1 * inv * g[c1] + bta[c1];
  xln[row * N_ + c0] = o0;  xln[row * N_ + c1] = o1;
  xh[row * N_ + c0] = (_Float16)o0;  xh[row * N_ + c1] = (_Float16)o1;
}

// per-(b,n) time mean and inverse centered norm over T
__global__ void __launch_bounds__(256) k_tstats(const float* __restrict__ xln,
                                                float* __restrict__ mean,
                                                float* __restrict__ invn) {
  __shared__ float sd[256];
  int b = blockIdx.x / N_, n = blockIdx.x % N_;
  float s = 0.f, ss = 0.f;
  for (int t = threadIdx.x; t < T_; t += 256) {
    float v = xln[((long long)b * T_ + t) * N_ + n];
    s += v; ss += v * v;
  }
  float S  = block_reduce_sum256(s, sd);
  float SS = block_reduce_sum256(ss, sd);
  if (threadIdx.x == 0) {
    float mu  = S * (1.0f / T_);
    float nrm = sqrtf(fmaxf(SS - (float)T_ * mu * mu, 0.f));
    mean[blockIdx.x] = mu;
    invn[blockIdx.x] = 1.f / fmaxf(nrm, EPS_);
  }
}

// materialize xnT[b][n][t] = (xln[b][t][n]-mean)*invn as f16 (K-contiguous)
__global__ void __launch_bounds__(256) k_xnt(const float* __restrict__ xln,
                                             const float* __restrict__ mean,
                                             const float* __restrict__ invn,
                                             _Float16* __restrict__ xnT) {
  int b = blockIdx.x / N_, n = blockIdx.x % N_;
  float mu = mean[blockIdx.x], iv = invn[blockIdx.x];
  long long obase = ((long long)b * N_ + n) * T_;
  for (int t = threadIdx.x; t < T_; t += 256) {
    float v = xln[((long long)b * T_ + t) * N_ + n];
    xnT[obase + t] = (_Float16)((v - mu) * iv);
  }
}

// mixed = a*tanh(0.5*(W+W^T)) + (1-a)*clip(corr); also A0 (zero diag)
__global__ void __launch_bounds__(256) k_mix(const float* __restrict__ corr,
                                             const float* __restrict__ Wc,
                                             const float* __restrict__ alpha,
                                             float* __restrict__ mixed_out,
                                             float* __restrict__ A0) {
  int i = blockIdx.x * 256 + threadIdx.x;
  if (i >= N_ * N_) return;
  int n = i / N_, m = i % N_;
  float a = 1.f / (1.f + expf(-alpha[0]));
  float learned = tanhf(0.5f * (Wc[i] + Wc[m * N_ + n]));
  float dc = clipf(corr[i], -1.f, 1.f);
  float mx = a * learned + (1.f - a) * dc;
  mixed_out[i] = mx;
  A0[i] = (n == m) ? 0.f : mx;
}

__global__ void __launch_bounds__(256) k_deg(const float* __restrict__ A0,
                                             float* __restrict__ dis) {
  __shared__ float sd[256];
  int n = blockIdx.x;
  float s = A0[(long long)n * N_ + threadIdx.x] + A0[(long long)n * N_ + threadIdx.x + 256];
  float S = block_reduce_sum256(s, sd);
  if (threadIdx.x == 0) dis[n] = rsqrtf(fmaxf(S, EPS_));
}

// Laplacian, stored TRANSPOSED: LhT[m][n] = clip(I - D^-1/2 A D^-1/2, +-2)[n][m]
__global__ void __launch_bounds__(256) k_lap(const float* __restrict__ A0,
                                             const float* __restrict__ dis,
                                             _Float16* __restrict__ LhT) {
  int i = blockIdx.x * 256 + threadIdx.x;
  if (i >= N_ * N_) return;
  int n = i / N_, m = i % N_;
  float L = ((n == m) ? 1.f : 0.f) - dis[n] * A0[i] * dis[m];
  LhT[(long long)m * N_ + n] = (_Float16)clipf(L, -2.f, 2.f);
}

// LN over H then exact GELU, writes f16
__global__ void __launch_bounds__(256) k_ln_gelu(const float* __restrict__ hpre,
                                                 const float* __restrict__ g,
                                                 const float* __restrict__ bta,
                                                 _Float16* __restrict__ hh) {
  __shared__ float sd[256];
  long long row = blockIdx.x;
  float v[4];
  float s = 0.f;
  #pragma unroll
  for (int j = 0; j < 4; ++j) { v[j] = hpre[row * H_ + threadIdx.x + j * 256]; s += v[j]; }
  float mu = block_reduce_sum256(s, sd) * (1.0f / H_);
  float q = 0.f;
  #pragma unroll
  for (int j = 0; j < 4; ++j) { v[j] -= mu; q += v[j] * v[j]; }
  float inv = rsqrtf(block_reduce_sum256(q, sd) * (1.0f / H_) + LN_EPS_);
  #pragma unroll
  for (int j = 0; j < 4; ++j) {
    int c = threadIdx.x + j * 256;
    float o = v[j] * inv * g[c] + bta[c];
    float ge = 0.5f * o * (1.f + erff(o * 0.70710678118654752f));
    hh[row * H_ + c] = (_Float16)ge;
  }
}

// final LN over O in-place in d_out
__global__ void __launch_bounds__(256) k_ln_out(float* __restrict__ y,
                                                const float* __restrict__ g,
                                                const float* __restrict__ bta) {
  __shared__ float sd[256];
  long long row = blockIdx.x;
  int c0 = threadIdx.x, c1 = threadIdx.x + 256;
  float v0 = y[row * O_ + c0], v1 = y[row * O_ + c1];
  float mu = block_reduce_sum256(v0 + v1, sd) * (1.0f / O_);
  float d0 = v0 - mu, d1 = v1 - mu;
  float inv = rsqrtf(block_reduce_sum256(d0 * d0 + d1 * d1, sd) * (1.0f / O_) + LN_EPS_);
  y[row * O_ + c0] = d0 * inv * g[c0] + bta[c0];
  y[row * O_ + c1] = d1 * inv * g[c1] + bta[c1];
}

// ---------------- WMMA GEMM kernels ----------------

// corr[n,m] = (1/B) * sum_b sum_t xnT[b,n,t]*xnT[b,m,t]  (M=N=512, K=B*T)
__global__ void __launch_bounds__(256, 1)
k_corr_gemm(const _Float16* __restrict__ xnT, float* __restrict__ corr) {
  __shared__ _Float16 As[2][BM][LDK_S];
  __shared__ _Float16 Bs[2][BN][LDK_S];
  int rowBase = blockIdx.y * BM, colBase = blockIdx.x * BN;
  int lane = threadIdx.x & 31, wave = threadIdx.x >> 5;
  int wm = wave & 3, wn = wave >> 2;
  v8f acc[2][4] = {};
  gemm_core(xnT, T_, xnT, T_, T_, B_, (long long)N_ * T_, (long long)N_ * T_,
            rowBase, colBase, As, Bs, acc);
  int hi = (lane >> 4) & 1, ln = lane & 15;
  #pragma unroll
  for (int tm = 0; tm < 2; ++tm)
    #pragma unroll
    for (int tn = 0; tn < 4; ++tn)
      #pragma unroll
      for (int r = 0; r < 8; ++r) {
        int row = rowBase + wm * 32 + tm * 16 + r + 8 * hi;
        int col = colBase + wn * 64 + tn * 16 + ln;
        corr[(long long)row * N_ + col] = acc[tm][tn][r] * (1.0f / B_);
      }
}

// generic GEMM: C = A[M,K] * Bt[N,K]^T, epilogue by mode:
//   mode 0: outH = (f16)acc                         (Tx1)
//   mode 1: Chebyshev combine with xln/tx1h/chebW   -> outH
//   mode 2: outF = acc + bias[col]                  (MLP pre-activations)
__global__ void __launch_bounds__(256, 1)
k_gemm(const _Float16* __restrict__ A, const _Float16* __restrict__ Bt,
       int K, int Nout, int mode,
       const float* __restrict__ bias, const float* __restrict__ xln,
       const _Float16* __restrict__ tx1h, const float* __restrict__ chebW,
       float* __restrict__ outF, _Float16* __restrict__ outH) {
  __shared__ _Float16 As[2][BM][LDK_S];
  __shared__ _Float16 Bs[2][BN][LDK_S];
  int rowBase = blockIdx.y * BM, colBase = blockIdx.x * BN;
  int lane = threadIdx.x & 31, wave = threadIdx.x >> 5;
  int wm = wave & 3, wn = wave >> 2;
  v8f acc[2][4] = {};
  gemm_core(A, K, Bt, K, K, 1, 0, 0, rowBase, colBase, As, Bs, acc);

  float w0 = 0.f, w1 = 0.f, w2 = 0.f;
  if (mode == 1) {
    float c0 = chebW[0], c1 = chebW[1], c2 = chebW[2];
    float mx = fmaxf(c0, fmaxf(c1, c2));
    float e0 = expf(c0 - mx), e1 = expf(c1 - mx), e2 = expf(c2 - mx);
    float inv = 1.f / (e0 + e1 + e2);
    w0 = e0 * inv; w1 = e1 * inv; w2 = e2 * inv;
  }
  int hi = (lane >> 4) & 1, ln = lane & 15;
  #pragma unroll
  for (int tm = 0; tm < 2; ++tm)
    #pragma unroll
    for (int tn = 0; tn < 4; ++tn)
      #pragma unroll
      for (int r = 0; r < 8; ++r) {
        long long row = rowBase + wm * 32 + tm * 16 + r + 8 * hi;
        int col = colBase + wn * 64 + tn * 16 + ln;
        long long idx = row * Nout + col;
        float v = acc[tm][tn][r];
        if (mode == 0) {
          outH[idx] = (_Float16)v;
        } else if (mode == 1) {
          float xv = xln[idx];
          float t1 = (float)tx1h[idx];
          float t2 = clipf(2.f * v - xv, -100.f, 100.f);
          outH[idx] = (_Float16)(w0 * xv + w1 * t1 + w2 * t2);
        } else {
          outF[idx] = v + bias[col];
        }
      }
}

// ---------------- host launch ----------------
extern "C" void kernel_launch(void* const* d_in, const int* in_sizes, int n_in,
                              void* d_out, int out_size, void* d_ws, size_t ws_size,
                              hipStream_t stream) {
  const float* x      = (const float*)d_in[0];
  const float* Wc     = (const float*)d_in[1];
  const float* alpha  = (const float*)d_in[2];
  const float* chebW  = (const float*)d_in[3];
  const float* Wi     = (const float*)d_in[4];
  const float* bi     = (const float*)d_in[5];
  const float* Wo     = (const float*)d_in[6];
  const float* bo     = (const float*)d_in[7];
  const float* g1 = (const float*)d_in[8],  *b1 = (const float*)d_in[9];
  const float* g2 = (const float*)d_in[10], *b2 = (const float*)d_in[11];
  const float* g3 = (const float*)d_in[12], *b3 = (const float*)d_in[13];
  float* y     = (float*)d_out;                                  // [M, O]
  float* mixed = (float*)d_out + (size_t)M_ * O_;                // [N, N]

  char* ws = (char*)d_ws;
  float*    xln  = (float*)(ws + XLN_OFF);
  _Float16* xh   = (_Float16*)(ws + XH_OFF);
  _Float16* xnT  = (_Float16*)(ws + XNT_OFF);
  float*    hpre = (float*)(ws + HPRE_OFF);
  _Float16* WiH  = (_Float16*)(ws + WIH_OFF);
  _Float16* WoH  = (_Float16*)(ws + WOH_OFF);
  _Float16* LhT  = (_Float16*)(ws + LHT_OFF);
  float*    corr = (float*)(ws + CORR_OFF);
  float*    A0   = (float*)(ws + A0_OFF);
  float*    mean = (float*)(ws + MEAN_OFF);
  float*    invn = (float*)(ws + INVN_OFF);
  float*    dis  = (float*)(ws + DIS_OFF);
  // region reuse (lifetimes are disjoint):
  _Float16* tx1h = xnT;                         // xnT dead after corr GEMM
  _Float16* outh = xh;                          // xh dead after Tx1 GEMM
  _Float16* hh   = (_Float16*)(ws + XLN_OFF);   // xln dead after Cheb epilogue

  // 1) cast weights to f16 (already in Bt = [out,in] layout)
  k_cast_f16<<<(H_ * N_ + 255) / 256, 256, 0, stream>>>(Wi, WiH, H_ * N_);
  k_cast_f16<<<(O_ * H_ + 255) / 256, 256, 0, stream>>>(Wo, WoH, O_ * H_);

  // 2) LN over features
  k_ln_x<<<(int)M_, 256, 0, stream>>>(x, g1, b1, xln, xh);

  // 3) time stats + normalized transposed copy
  k_tstats<<<B_ * N_, 256, 0, stream>>>(xln, mean, invn);
  k_xnt<<<B_ * N_, 256, 0, stream>>>(xln, mean, invn, xnT);

  // 4) correlation GEMM (WMMA), mean over batch folded into epilogue
  k_corr_gemm<<<dim3(N_ / BN, N_ / BM), 256, 0, stream>>>(xnT, corr);

  // 5) mix, degree, Laplacian (transposed f16)
  k_mix<<<(N_ * N_ + 255) / 256, 256, 0, stream>>>(corr, Wc, alpha, mixed, A0);
  k_deg<<<N_, 256, 0, stream>>>(A0, dis);
  k_lap<<<(N_ * N_ + 255) / 256, 256, 0, stream>>>(A0, dis, LhT);

  // 6) Tx1 = xln @ L   (mode 0 -> f16)
  k_gemm<<<dim3(N_ / BN, (int)(M_ / BM)), 256, 0, stream>>>(
      xh, LhT, N_, N_, 0, nullptr, nullptr, nullptr, nullptr, nullptr, tx1h);

  // 7) Tx2 + Chebyshev combine (mode 1 -> f16 "out")
  k_gemm<<<dim3(N_ / BN, (int)(M_ / BM)), 256, 0, stream>>>(
      tx1h, LhT, N_, N_, 1, nullptr, xln, tx1h, chebW, nullptr, outh);

  // 8) h = out @ Wi^T + bi  (mode 2 -> f32 hpre)
  k_gemm<<<dim3(H_ / BN, (int)(M_ / BM)), 256, 0, stream>>>(
      outh, WiH, N_, H_, 2, bi, nullptr, nullptr, nullptr, hpre, nullptr);

  // 9) LN + GELU over H -> f16
  k_ln_gelu<<<(int)M_, 256, 0, stream>>>(hpre, g2, b2, hh);

  // 10) y = hh @ Wo^T + bo  (mode 2 -> f32 into d_out)
  k_gemm<<<dim3(O_ / BN, (int)(M_ / BM)), 256, 0, stream>>>(
      hh, WoH, H_, O_, 2, bo, nullptr, nullptr, nullptr, y, nullptr);

  // 11) final LN over O, in place
  k_ln_out<<<(int)M_, 256, 0, stream>>>(y, g3, b3);
}
